// MultiHeadAttention_77481210020030
// MI455X (gfx1250) — matrix operand order repaired and checked by reading, not compiled, not run on hardware
//
#include <hip/hip_runtime.h>

// ---------------------------------------------------------------------------
// MI455X (gfx1250) multi-head attention, bf16 WMMA, f32 accumulate.
// Round 3:
//  - 1/sqrt(dk) folded into the Q projection epilogue (no per-tile scaling)
//  - GEMM tiles 128x64: each wave owns two 16-row groups -> B-frag reuse,
//    8 WMMAs per k-step per wave
//  - K/Q attention staging via GLOBAL_LOAD_ASYNC_TO_LDS_B128 (inline asm,
//    ASYNCcnt-tracked CDNA5 async DMA), V staged transposed via vector loads
// Fragment layouts per CDNA5 ISA 7.12.2 (wave32):
//   A 16x32 bf16: lane L row M=L&15; elem i<8 -> K=i+8*(L>=16); i>=8 -> K=16+(i-8)+8*(L>=16)
//   B 32x16 bf16: lane L col N=L&15; elem i<8 -> K=i+8*(L>=16); i>=8 -> K=16+(i-8)+8*(L>=16)
//   C/D 16x16 f32: lane L col N=L&15; VGPR v -> row M=v+8*(L>=16)
// ---------------------------------------------------------------------------

typedef __bf16 bf16;
typedef __attribute__((ext_vector_type(16))) __bf16 v16bf;
typedef __attribute__((ext_vector_type(8)))  __bf16 bf16x8;
typedef __attribute__((ext_vector_type(4)))  __bf16 bf16x4;
typedef __attribute__((ext_vector_type(8)))  float  v8f;

#define D_MODEL 1024
#define N_HEADS 16
#define DK      64
#define BATCH   4
#define SEQ     2048
#define MTOT    (BATCH * SEQ)   // 8192
#define LDP     72              // padded LDS row stride (64 + 8) bf16 elems

// ---- staging helpers ------------------------------------------------------
__device__ inline bf16x4 load4(const float* p) {
    const float4 f = *(const float4*)p;
    bf16x4 r; r[0] = (bf16)f.x; r[1] = (bf16)f.y; r[2] = (bf16)f.z; r[3] = (bf16)f.w;
    return r;
}
__device__ inline bf16x4 load4(const bf16* p) { return *(const bf16x4*)p; }

// build a 16-wide fragment from two aligned 8-element LDS runs
__device__ inline v16bf frag16(const bf16* p0, const bf16* p1) {
    bf16x8 lo = *(const bf16x8*)p0;
    bf16x8 hi = *(const bf16x8*)p1;
    return __builtin_shufflevector(lo, hi, 0, 1, 2, 3, 4, 5, 6, 7,
                                            8, 9, 10, 11, 12, 13, 14, 15);
}

// CDNA5 async global->LDS DMA, 16 bytes per lane (GV mode: VDST = LDS byte
// address VGPR, VADDR = 64-bit global address). Tracked with ASYNCcnt.
__device__ inline void async_copy16(bf16* lds_dst, const bf16* gsrc) {
    unsigned lds_off = (unsigned)(unsigned long long)lds_dst;   // low 32b = LDS addr
    asm volatile("global_load_async_to_lds_b128 %0, %1, off"
                 :: "v"(lds_off), "v"((unsigned long long)gsrc)
                 : "memory");
}
__device__ inline void async_wait0() {
    asm volatile("s_wait_asynccnt 0x0" ::: "memory");
}

// ---------------------------------------------------------------------------
// GEMM: Y[M,N] = (A[M,K] @ W[N,K]^T + bias[N]) * oscale
// 128x64 tile, 4 waves (128 thr), K-stage 64 = two WMMA k-steps.
// Each wave owns two 16-row groups (wv*16 and 64+wv*16): B-frags reused 2x.
// ---------------------------------------------------------------------------
template <typename AT, typename OT>
__global__ __launch_bounds__(128)
void gemm_bias_wmma(const AT* __restrict__ A, const float* __restrict__ W,
                    const float* __restrict__ bias, OT* __restrict__ Y,
                    int M, int N, int K, float oscale)
{
    __shared__ bf16 x_lds[128][LDP];  // A tile [m][k]   (18 KB)
    __shared__ bf16 w_lds[64][LDP];   // W tile [n][k]   ( 9 KB)

    const int tid  = threadIdx.x;
    const int lane = tid & 31;
    const int wv   = tid >> 5;
    const int row  = lane & 15;
    const int hi   = lane >> 4;

    const int m0 = blockIdx.x * 128;
    const int n0 = blockIdx.y * 64;

    v8f acc[2][4];
    #pragma unroll
    for (int g = 0; g < 2; ++g)
        #pragma unroll
        for (int j = 0; j < 4; ++j)
            #pragma unroll
            for (int v = 0; v < 8; ++v) acc[g][j][v] = 0.0f;

    for (int k0 = 0; k0 < K; k0 += 64) {
        __syncthreads();
        #pragma unroll
        for (int it = 0; it < 16; ++it) {          // A tile: 128x64
            int e = (tid + it * 128) * 4;          // 0..8191
            int r = e >> 6, c = e & 63;
            *(bf16x4*)&x_lds[r][c] = load4(&A[(size_t)(m0 + r) * K + k0 + c]);
        }
        #pragma unroll
        for (int it = 0; it < 8; ++it) {           // W tile: 64x64
            int e = (tid + it * 128) * 4;          // 0..4095
            int r = e >> 6, c = e & 63;
            *(bf16x4*)&w_lds[r][c] = load4(&W[(size_t)(n0 + r) * K + k0 + c]);
        }
        if (k0 + 64 < K) {
            __builtin_prefetch(&A[(size_t)(m0 + tid) * K + k0 + 64], 0, 1);
            __builtin_prefetch(&W[(size_t)(n0 + (tid >> 1)) * K + k0 + 64], 0, 1);
        }
        __syncthreads();

        #pragma unroll
        for (int ks = 0; ks < 2; ++ks) {
            const bf16* a0r = &x_lds[wv * 16 + row][ks * 32];
            const bf16* a1r = &x_lds[64 + wv * 16 + row][ks * 32];
            v16bf a0 = frag16(a0r + 8 * hi, a0r + 16 + 8 * hi);
            v16bf a1 = frag16(a1r + 8 * hi, a1r + 16 + 8 * hi);
            #pragma unroll
            for (int j = 0; j < 4; ++j) {
                const bf16* br = &w_lds[j * 16 + row][ks * 32 + 8 * hi];
                v16bf b = frag16(br, br + 16);
                acc[0][j] = __builtin_amdgcn_wmma_f32_16x16x32_bf16(
                                false, a0, false, b, (short)0, acc[0][j], false, false);
                acc[1][j] = __builtin_amdgcn_wmma_f32_16x16x32_bf16(
                                false, a1, false, b, (short)0, acc[1][j], false, false);
            }
        }
    }

    #pragma unroll
    for (int g = 0; g < 2; ++g) {
        #pragma unroll
        for (int j = 0; j < 4; ++j) {
            int n = n0 + j * 16 + row;
            float bv = bias[n];
            #pragma unroll
            for (int v = 0; v < 8; ++v) {
                int m = m0 + g * 64 + wv * 16 + v + 8 * hi;
                Y[(size_t)m * N + n] = (OT)((acc[g][j][v] + bv) * oscale);
            }
        }
    }
}

// ---------------------------------------------------------------------------
// Flash attention per (batch, head, 64-query tile); 4 waves x 16 rows.
// Q is pre-scaled by 1/sqrt(dk) in its projection. Online softmax fully in
// registers; per-row reductions are 4-step shfl_xor butterflies inside the
// 16-lane half that owns the row's columns.
// ---------------------------------------------------------------------------
__global__ __launch_bounds__(128)
void attention_wmma(const bf16* __restrict__ Qb, const bf16* __restrict__ Kb,
                    const bf16* __restrict__ Vb, bf16* __restrict__ Cb)
{
    __shared__ bf16 q_lds[64][LDP];        // Q tile  [q][d]
    __shared__ bf16 k_lds[64][LDP];        // K tile  [key][d]
    __shared__ bf16 vt_lds[64][LDP];       // V^T tile [d][key]
    __shared__ bf16 p_lds[4][16][LDP];     // per-wave P tile [row][key]

    const int tid  = threadIdx.x;
    const int lane = tid & 31;
    const int wv   = tid >> 5;
    const int row  = lane & 15;
    const int hi   = lane >> 4;

    const int q0 = blockIdx.x * 64;
    const int h  = blockIdx.y;
    const int b  = blockIdx.z;
    const size_t base = ((size_t)b * SEQ) * D_MODEL + (size_t)h * DK;

    // ---- async DMA the Q tile into LDS ----
    #pragma unroll
    for (int it = 0; it < 4; ++it) {
        int e = (tid + it * 128) * 8;              // 0..4095
        int r = e >> 6, d = e & 63;
        async_copy16(&q_lds[r][d], &Qb[base + (size_t)(q0 + r) * D_MODEL + d]);
    }
    async_wait0();
    __syncthreads();

    v16bf qf[2];
    #pragma unroll
    for (int ks = 0; ks < 2; ++ks) {
        const bf16* qr = &q_lds[wv * 16 + row][ks * 32];
        qf[ks] = frag16(qr + 8 * hi, qr + 16 + 8 * hi);
    }

    v8f of[4];
    #pragma unroll
    for (int j = 0; j < 4; ++j)
        #pragma unroll
        for (int v = 0; v < 8; ++v) of[j][v] = 0.0f;

    float m_state[8], l_state[8];
    #pragma unroll
    for (int v = 0; v < 8; ++v) { m_state[v] = -1e30f; l_state[v] = 0.0f; }

    for (int kt = 0; kt < SEQ / 64; ++kt) {
        const int kb0 = kt * 64;
        __syncthreads();                            // protect k/vt/p reuse
        // ---- K: async DMA; V: vector load + transpose into LDS ----
        #pragma unroll
        for (int it = 0; it < 4; ++it) {
            int e = (tid + it * 128) * 8;
            int r = e >> 6, d = e & 63;
            async_copy16(&k_lds[r][d], &Kb[base + (size_t)(kb0 + r) * D_MODEL + d]);
            bf16x8 vv = *(const bf16x8*)&Vb[base + (size_t)(kb0 + r) * D_MODEL + d];
            #pragma unroll
            for (int q = 0; q < 8; ++q) vt_lds[d + q][r] = vv[q];
        }
        async_wait0();
        __syncthreads();

        // ---- scores: S = Q (16x64) @ K^T, 4 key-tiles x 2 k-steps ----
        v8f sf[4];
        #pragma unroll
        for (int j = 0; j < 4; ++j) {
            #pragma unroll
            for (int v = 0; v < 8; ++v) sf[j][v] = 0.0f;
            #pragma unroll
            for (int ks = 0; ks < 2; ++ks) {
                const bf16* br = &k_lds[j * 16 + row][ks * 32 + 8 * hi];
                v16bf bb = frag16(br, br + 16);
                sf[j] = __builtin_amdgcn_wmma_f32_16x16x32_bf16(
                            false, qf[ks], false, bb, (short)0, sf[j], false, false);
            }
        }

        // ---- online softmax in registers ----
        float corr_v[8];
        #pragma unroll
        for (int v = 0; v < 8; ++v) {
            float mx = fmaxf(fmaxf(sf[0][v], sf[1][v]), fmaxf(sf[2][v], sf[3][v]));
            mx = fmaxf(mx, __shfl_xor(mx, 1, 32));
            mx = fmaxf(mx, __shfl_xor(mx, 2, 32));
            mx = fmaxf(mx, __shfl_xor(mx, 4, 32));
            mx = fmaxf(mx, __shfl_xor(mx, 8, 32));
            float mnew = fmaxf(m_state[v], mx);
            float corr = __expf(m_state[v] - mnew);
            m_state[v] = mnew;
            float ps = 0.0f;
            #pragma unroll
            for (int j = 0; j < 4; ++j) {
                float p = __expf(sf[j][v] - mnew);
                sf[j][v] = p;                       // reuse sf as P
                ps += p;
            }
            ps += __shfl_xor(ps, 1, 32);
            ps += __shfl_xor(ps, 2, 32);
            ps += __shfl_xor(ps, 4, 32);
            ps += __shfl_xor(ps, 8, 32);
            l_state[v] = l_state[v] * corr + ps;
            corr_v[v]  = corr;
        }

        // ---- write P (C-layout -> LDS) and rescale O ----
        #pragma unroll
        for (int j = 0; j < 4; ++j)
            #pragma unroll
            for (int v = 0; v < 8; ++v)
                p_lds[wv][v + 8 * hi][j * 16 + row] = (bf16)sf[j][v];
        #pragma unroll
        for (int j = 0; j < 4; ++j)
            #pragma unroll
            for (int v = 0; v < 8; ++v) of[j][v] *= corr_v[v];
        __syncthreads();

        // ---- O += P (16x64) @ V (64x64) ----
        #pragma unroll
        for (int ks = 0; ks < 2; ++ks) {
            const bf16* pr = &p_lds[wv][row][ks * 32];
            v16bf pa = frag16(pr + 8 * hi, pr + 16 + 8 * hi);
            #pragma unroll
            for (int j = 0; j < 4; ++j) {
                const bf16* vr = &vt_lds[j * 16 + row][ks * 32 + 8 * hi];
                v16bf vb = frag16(vr, vr + 16);
                of[j] = __builtin_amdgcn_wmma_f32_16x16x32_bf16(
                            false, pa, false, vb, (short)0, of[j], false, false);
            }
        }
    }

    // ---- normalize (per-lane row state) and write ctx ----
    #pragma unroll
    for (int v = 0; v < 8; ++v) {
        float linv = 1.0f / l_state[v];
        int s = q0 + wv * 16 + v + 8 * hi;
        #pragma unroll
        for (int j = 0; j < 4; ++j) {
            int d = j * 16 + row;
            Cb[base + (size_t)s * D_MODEL + d] = (bf16)(of[j][v] * linv);
        }
    }
}

// ---------------------------------------------------------------------------
// Launch: 3 projection GEMMs -> attention -> output GEMM. All on `stream`.
// Workspace: 4 x (8192*1024) bf16 = 64 MB (Qb, Kb, Vb, ctx).
// ---------------------------------------------------------------------------
extern "C" void kernel_launch(void* const* d_in, const int* in_sizes, int n_in,
                              void* d_out, int out_size, void* d_ws, size_t ws_size,
                              hipStream_t stream)
{
    const float* query  = (const float*)d_in[0];
    const float* key_in = (const float*)d_in[1];
    const float* value  = (const float*)d_in[2];
    const float* Wq = (const float*)d_in[3];
    const float* bq = (const float*)d_in[4];
    const float* Wk = (const float*)d_in[5];
    const float* bk = (const float*)d_in[6];
    const float* Wv = (const float*)d_in[7];
    const float* bv = (const float*)d_in[8];
    const float* Wo = (const float*)d_in[9];
    const float* bo = (const float*)d_in[10];
    float* out = (float*)d_out;

    const size_t elems = (size_t)MTOT * D_MODEL;
    bf16* Qb = (bf16*)d_ws;
    bf16* Kb = Qb + elems;
    bf16* Vb = Kb + elems;
    bf16* Cb = Vb + elems;

    dim3 gblk(128);
    dim3 ggrid(MTOT / 128, D_MODEL / 64);   // 64 x 16 workgroups per GEMM

    gemm_bias_wmma<float, bf16><<<ggrid, gblk, 0, stream>>>(query,  Wq, bq, Qb,
                                                            MTOT, D_MODEL, D_MODEL, 0.125f);
    gemm_bias_wmma<float, bf16><<<ggrid, gblk, 0, stream>>>(key_in, Wk, bk, Kb,
                                                            MTOT, D_MODEL, D_MODEL, 1.0f);
    gemm_bias_wmma<float, bf16><<<ggrid, gblk, 0, stream>>>(value,  Wv, bv, Vb,
                                                            MTOT, D_MODEL, D_MODEL, 1.0f);

    attention_wmma<<<dim3(SEQ / 64, N_HEADS, BATCH), gblk, 0, stream>>>(Qb, Kb, Vb, Cb);

    gemm_bias_wmma<bf16, float><<<ggrid, gblk, 0, stream>>>(Cb, Wo, bo, out,
                                                            MTOT, D_MODEL, D_MODEL, 1.0f);
}
